// RelativeMultiHeadAttention_66760971649650
// MI455X (gfx1250) — matrix operand order repaired and checked by reading, not compiled, not run on hardware
//
#include <hip/hip_runtime.h>
#include <hip/hip_bf16.h>
#include <stdint.h>
#include <stddef.h>

// Problem constants (match reference)
static constexpr int Bc  = 4;
static constexpr int SQc = 512;
static constexpr int SKc = 512;
static constexpr int Dc  = 512;
static constexpr int Hc  = 8;
static constexpr int DHc = 64;

typedef __attribute__((ext_vector_type(16))) __bf16 bf16x16;
typedef __attribute__((ext_vector_type(8)))  __bf16 bf16x8;
typedef __attribute__((ext_vector_type(2)))  __bf16 bf16x2;
typedef __attribute__((ext_vector_type(8)))  float  f32x8;
typedef __attribute__((ext_vector_type(4)))  float  f32x4;

// ---------------------------------------------------------------------------
// WMMA fragment helpers (CDNA5 wave32 layouts, CDNA5 ISA §7.12.2)
// A 16x32 bf16: lane L holds row M=L%16; VGPR j holds K = (L/16)*16 + j*2 (+1), contiguous like B
// B 32x16 bf16 (stored as Bt row-major, row = N, contiguous K): lane L holds
//   col N=L%16, elements K = k0 + (L/16)*16 + e, contiguous.
// C/D f32 16x16: VGPR j -> M = j + (L/16)*8, N = L%16.
// ---------------------------------------------------------------------------

static __device__ __forceinline__ bf16x16 load_a_frag(const __bf16* rowptr, int lane) {
  const int hi16 = (lane >> 4) << 4;
  bf16x16 a;
#pragma unroll
  for (int j = 0; j < 8; ++j) {
    const int k = hi16 + (j << 1);
    bf16x2 p = *(const bf16x2*)(rowptr + k);
    a[2 * j]     = p.x;
    a[2 * j + 1] = p.y;
  }
  return a;
}

static __device__ __forceinline__ bf16x16 load_b_frag(const __bf16* p16) {
  const bf16x8* p = (const bf16x8*)p16;
  bf16x8 lo = p[0];
  bf16x8 hi = p[1];
  bf16x16 b;
#pragma unroll
  for (int j = 0; j < 8; ++j) { b[j] = lo[j]; b[8 + j] = hi[j]; }
  return b;
}

static __device__ __forceinline__ bf16x16 load_b_frag_f32(const float* p32) {
  bf16x16 b;
#pragma unroll
  for (int j = 0; j < 4; ++j) {
    f32x4 v = *(const f32x4*)(p32 + 4 * j);
    b[4 * j + 0] = (__bf16)v.x;
    b[4 * j + 1] = (__bf16)v.y;
    b[4 * j + 2] = (__bf16)v.z;
    b[4 * j + 3] = (__bf16)v.w;
  }
  return b;
}

static __device__ __forceinline__ f32x8 wmma_bf16(bf16x16 a, bf16x16 b, f32x8 c) {
  // v_wmma_f32_16x16x32_bf16 : D = A x B + C, f32 accumulate
  return __builtin_amdgcn_wmma_f32_16x16x32_bf16(false, a, false, b, (short)0, c,
                                                 false, false);
}

// ---------------------------------------------------------------------------
// Generic batched bf16 WMMA GEMM:  C[z] = A[z] @ Bt[z]^T (+bias) (+residual)
// Bt is B transposed (row = N, contiguous K) so B-fragments are contiguous.
// Tile: BM=64, BN=64, BK=32.  128 threads = 4 waves; wave w owns rows w*16..+15
// and all four 16-col tiles. LDS staging, row stride 40 (16B-aligned rows).
// z decomposed as (zb, zh) = (z / zH, z % zH); per-axis pointer offsets.
// ---------------------------------------------------------------------------
__global__ __launch_bounds__(128) void gemm_bf16_wmma(
    const __bf16* __restrict__ A, const __bf16* __restrict__ Bt,
    const float* __restrict__ bias, const float* __restrict__ residual,
    void* __restrict__ Cv, int c_is_bf16,
    int M, int N, int K, int lda, int ldb, int ldc,
    int zH, long zAb, long zAh, long zBb, long zBh, long zCb, long zCh) {
  (void)M; (void)N;
  __shared__ __bf16 lA[64 * 40];
  __shared__ __bf16 lB[64 * 40];

  const int z  = blockIdx.z;
  const int zb = z / zH;
  const int zh = z % zH;
  const __bf16* Ab = A  + (size_t)zAb * zb + (size_t)zAh * zh;
  const __bf16* Bb = Bt + (size_t)zBb * zb + (size_t)zBh * zh;

  const int n0   = blockIdx.x * 64;
  const int m0   = blockIdx.y * 64;
  const int t    = threadIdx.x;
  const int w    = t >> 5;
  const int lane = t & 31;
  const int lrow = lane & 15;
  const int hi   = lane >> 4;

  const f32x8 fzero = {0.f, 0.f, 0.f, 0.f, 0.f, 0.f, 0.f, 0.f};
  f32x8 acc[4];
#pragma unroll
  for (int nt = 0; nt < 4; ++nt) acc[nt] = fzero;

  const int ldl   = 40;
  const int trow  = t >> 1;          // 0..63
  const int thalf = (t & 1) * 16;    // 0 or 16

  for (int kk = 0; kk < K; kk += 32) {
    const __bf16* ga = Ab + (size_t)(m0 + trow) * lda + kk + thalf;
    *(bf16x8*)(lA + trow * ldl + thalf)     = *(const bf16x8*)(ga);
    *(bf16x8*)(lA + trow * ldl + thalf + 8) = *(const bf16x8*)(ga + 8);
    const __bf16* gb = Bb + (size_t)(n0 + trow) * ldb + kk + thalf;
    *(bf16x8*)(lB + trow * ldl + thalf)     = *(const bf16x8*)(gb);
    *(bf16x8*)(lB + trow * ldl + thalf + 8) = *(const bf16x8*)(gb + 8);
    __syncthreads();

    const bf16x16 af = load_a_frag(lA + (w * 16 + lrow) * ldl, lane);
#pragma unroll
    for (int nt = 0; nt < 4; ++nt) {
      const bf16x16 bfv = load_b_frag(lB + (nt * 16 + lrow) * ldl + (hi << 4));
      acc[nt] = wmma_bf16(af, bfv, acc[nt]);
    }
    __syncthreads();
  }

  const int mrow0 = m0 + w * 16 + hi * 8;
  if (c_is_bf16) {
    __bf16* C = (__bf16*)Cv + (size_t)zCb * zb + (size_t)zCh * zh;
#pragma unroll
    for (int nt = 0; nt < 4; ++nt) {
      const int cn = n0 + nt * 16 + lrow;
      const float bv = bias ? bias[cn] : 0.f;
#pragma unroll
      for (int j = 0; j < 8; ++j) {
        C[(size_t)(mrow0 + j) * ldc + cn] = (__bf16)(acc[nt][j] + bv);
      }
    }
  } else {
    float* C = (float*)Cv + (size_t)zCb * zb + (size_t)zCh * zh;
#pragma unroll
    for (int nt = 0; nt < 4; ++nt) {
      const int cn = n0 + nt * 16 + lrow;
      const float bv = bias ? bias[cn] : 0.f;
#pragma unroll
      for (int j = 0; j < 8; ++j) {
        float v = acc[nt][j] + bv;
        if (residual) v += residual[(size_t)(mrow0 + j) * ldc + cn];
        C[(size_t)(mrow0 + j) * ldc + cn] = v;
      }
    }
  }
}

// ---------------------------------------------------------------------------
// bd kernel: score[b,h,q,k] += sum_d p[b,q,h,d] * pos_emb[q,k,d]
// Per block: one q, 64 keys. 8 waves: wave w -> (rowtile = w&1 of 32 (b,h)
// rows, keytile = w>>1). B-fragments stream pos_emb (f32 -> bf16 on load).
// ---------------------------------------------------------------------------
__global__ __launch_bounds__(256) void bd_kernel(
    const __bf16* __restrict__ p_bf, const float* __restrict__ pos_emb,
    float* __restrict__ score) {
  const int q    = blockIdx.y;
  const int w    = threadIdx.x >> 5;
  const int lane = threadIdx.x & 31;
  const int rt   = w & 1;         // row-tile over 32 (b,h) rows
  const int nt   = w >> 1;        // 4 key tiles of 16
  const int n0   = blockIdx.x * 64 + nt * 16;
  const int lrow = lane & 15;
  const int hi   = lane >> 4;

  const int rga = rt * 16 + lrow;            // (b*H + h) row for A
  const int ab  = rga >> 3;                  // / H
  const int ah  = rga & 7;                   // % H
  // p layout: [(b*SQ + q)*H + h][Dc]
  const __bf16* arow = p_bf + (((size_t)ab * SQc + q) * Hc + ah) * Dc;

  const int   kcol = n0 + lrow;
  const float* brow = pos_emb + ((size_t)q * SKc + kcol) * Dc + (hi << 4);

  const f32x8 fzero = {0.f, 0.f, 0.f, 0.f, 0.f, 0.f, 0.f, 0.f};
  f32x8 acc = fzero;
  for (int kk = 0; kk < Dc; kk += 32) {
    __builtin_prefetch(brow + kk + 128, 0, 1);   // stream pos_emb (read-once)
    const bf16x16 af  = load_a_frag(arow + kk, lane);
    const bf16x16 bfv = load_b_frag_f32(brow + kk);
    acc = wmma_bf16(af, bfv, acc);
  }

#pragma unroll
  for (int j = 0; j < 8; ++j) {
    const int rg = rt * 16 + hi * 8 + j;
    const int b  = rg >> 3;
    const int h  = rg & 7;
    const size_t idx = (((size_t)(b * Hc + h) * SQc) + q) * SKc + kcol;
    score[idx] += acc[j];
  }
}

// ---------------------------------------------------------------------------
// Elementwise / utility kernels
// ---------------------------------------------------------------------------
__global__ __launch_bounds__(256) void cvt_f32_bf16(const float* __restrict__ in,
                                                    __bf16* __restrict__ out, int n) {
  const int i = blockIdx.x * 256 + threadIdx.x;
  if (i < n) out[i] = (__bf16)in[i];
}

// Wt[j*D + i] = (bf16) W[i*D + j]   (32x32 LDS tiles)
__global__ __launch_bounds__(256) void transpose_cvt_w(const float* __restrict__ in,
                                                       __bf16* __restrict__ out) {
  __shared__ float tile[32][33];
  const int bx = blockIdx.x * 32;   // column block (j)
  const int by = blockIdx.y * 32;   // row block (i)
  const int tx = threadIdx.x & 31;
  const int ty = threadIdx.x >> 5;  // 0..7
#pragma unroll
  for (int r = 0; r < 32; r += 8)
    tile[ty + r][tx] = in[(size_t)(by + ty + r) * Dc + bx + tx];
  __syncthreads();
#pragma unroll
  for (int r = 0; r < 32; r += 8)
    out[(size_t)(bx + ty + r) * Dc + by + tx] = (__bf16)tile[tx][ty + r];
}

// qu = bf16(qh + u_bias), qv = bf16(qh + v_bias); biases indexed by col = h*DH+dh
__global__ __launch_bounds__(256) void addbias_uv(const float* __restrict__ qh,
                                                  const float* __restrict__ ub,
                                                  const float* __restrict__ vb,
                                                  __bf16* __restrict__ qu,
                                                  __bf16* __restrict__ qv, int n) {
  const int i = blockIdx.x * 256 + threadIdx.x;
  if (i >= n) return;
  const int c = i % Dc;
  const float v = qh[i];
  qu[i] = (__bf16)(v + ub[c]);
  qv[i] = (__bf16)(v + vb[c]);
}

// vT[((b*H+h)*DH + dh)*SK + k] = vh[(b*SK + k)*D + h*DH + dh]
__global__ __launch_bounds__(256) void transpose_v_kernel(const __bf16* __restrict__ vh,
                                                          __bf16* __restrict__ vT, int n) {
  const int i = blockIdx.x * 256 + threadIdx.x;
  if (i >= n) return;
  const int k  = i % SKc;
  const int r  = i / SKc;
  const int dh = r % DHc;
  const int bh = r / DHc;
  const int h  = bh % Hc;
  const int b  = bh / Hc;
  vT[i] = vh[((size_t)(b * SKc + k)) * Dc + h * DHc + dh];
}

// cadd[(b*SQ+q)*H + h] = sum_dh (qh + v_bias) * br   (the pos-bias br term)
__global__ __launch_bounds__(256) void brdot_kernel(const float* __restrict__ qh,
                                                    const float* __restrict__ vbias,
                                                    const float* __restrict__ br,
                                                    float* __restrict__ cadd, int n) {
  const int i = blockIdx.x * 256 + threadIdx.x;
  if (i >= n) return;
  const int h  = i % Hc;
  const int mq = i / Hc;   // b*SQ + q
  const float* qrow = qh + (size_t)mq * Dc + h * DHc;
  const float* vb   = vbias + h * DHc;
  const float* brh  = br + h * DHc;
  float s = 0.f;
#pragma unroll 8
  for (int d = 0; d < DHc; ++d) s += (qrow[d] + vb[d]) * brh[d];
  cadd[(size_t)mq * Hc + h] = s;
}

// Softmax over keys with scale, br-term, masks, all-masked -> 0.
__global__ __launch_bounds__(256) void softmax_kernel(
    const float* __restrict__ score, const float* __restrict__ cadd,
    const unsigned char* __restrict__ kpm, const unsigned char* __restrict__ amask,
    float* __restrict__ attn, __bf16* __restrict__ attn_bf) {
  __shared__ float red[256];
  const int row = blockIdx.x;                 // (b*H + h)*SQ + q
  const int q   = row % SQc;
  const int h   = (row / SQc) % Hc;
  const int b   = row / (SQc * Hc);
  const float ca = cadd[((size_t)b * SQc + q) * Hc + h];
  const int t = threadIdx.x;

  float s[2];
#pragma unroll
  for (int i = 0; i < 2; ++i) {
    const int k = t + i * 256;
    const float v = score[(size_t)row * SKc + k];
    const bool m = (kpm[b * SKc + k] != 0) || (amask[(size_t)q * SKc + k] != 0);
    s[i] = m ? -INFINITY : (v + ca) * 0.125f;   // 1/sqrt(64)
  }
  red[t] = fmaxf(s[0], s[1]);
  __syncthreads();
  for (int off = 128; off > 0; off >>= 1) {
    if (t < off) red[t] = fmaxf(red[t], red[t + off]);
    __syncthreads();
  }
  const float mx = red[0];
  __syncthreads();

  float e[2];
  if (mx == -INFINITY) { e[0] = 0.f; e[1] = 0.f; }
  else { e[0] = __expf(s[0] - mx); e[1] = __expf(s[1] - mx); }
  red[t] = e[0] + e[1];
  __syncthreads();
  for (int off = 128; off > 0; off >>= 1) {
    if (t < off) red[t] += red[t + off];
    __syncthreads();
  }
  const float sum = red[0];
  const float inv = (sum > 0.f) ? (1.f / sum) : 0.f;
#pragma unroll
  for (int i = 0; i < 2; ++i) {
    const int k = t + i * 256;
    const float wv = e[i] * inv;
    attn[(size_t)row * SKc + k]    = wv;
    attn_bf[(size_t)row * SKc + k] = (__bf16)wv;
  }
}

// LayerNorm over last dim (512) of o; writes normed output.
__global__ __launch_bounds__(256) void layernorm_kernel(const float* __restrict__ x,
                                                        const float* __restrict__ g,
                                                        const float* __restrict__ bta,
                                                        float* __restrict__ out) {
  __shared__ float red[256];
  const int row = blockIdx.x;
  const int t   = threadIdx.x;
  const float v0 = x[(size_t)row * Dc + t];
  const float v1 = x[(size_t)row * Dc + t + 256];
  red[t] = v0 + v1;
  __syncthreads();
  for (int off = 128; off > 0; off >>= 1) {
    if (t < off) red[t] += red[t + off];
    __syncthreads();
  }
  const float mean = red[0] * (1.f / Dc);
  __syncthreads();
  red[t] = v0 * v0 + v1 * v1;
  __syncthreads();
  for (int off = 128; off > 0; off >>= 1) {
    if (t < off) red[t] += red[t + off];
    __syncthreads();
  }
  const float var = red[0] * (1.f / Dc) - mean * mean;
  const float rs  = rsqrtf(var + 1e-5f);
  out[(size_t)row * Dc + t]       = (v0 - mean) * rs * g[t] + bta[t];
  out[(size_t)row * Dc + t + 256] = (v1 - mean) * rs * g[t + 256] + bta[t + 256];
}

// ---------------------------------------------------------------------------
// Host launcher
// ---------------------------------------------------------------------------
extern "C" void kernel_launch(void* const* d_in, const int* in_sizes, int n_in,
                              void* d_out, int out_size, void* d_ws, size_t ws_size,
                              hipStream_t stream) {
  (void)in_sizes; (void)n_in; (void)out_size; (void)ws_size;

  const float* q_in   = (const float*)d_in[0];
  const float* k_in   = (const float*)d_in[1];
  const float* v_in   = (const float*)d_in[2];
  const float* pos    = (const float*)d_in[3];
  const unsigned char* kpm   = (const unsigned char*)d_in[4];
  const unsigned char* amask = (const unsigned char*)d_in[5];
  const float* Wq = (const float*)d_in[6];  const float* bq = (const float*)d_in[7];
  const float* Wk = (const float*)d_in[8];  const float* bk = (const float*)d_in[9];
  const float* Wv = (const float*)d_in[10]; const float* bv = (const float*)d_in[11];
  const float* Wo = (const float*)d_in[12]; const float* bo = (const float*)d_in[13];
  const float* Wr = (const float*)d_in[14]; const float* br = (const float*)d_in[15];
  const float* ub = (const float*)d_in[16]; const float* vb = (const float*)d_in[17];
  const float* lg = (const float*)d_in[18]; const float* lb = (const float*)d_in[19];

  const size_t nQKV   = (size_t)Bc * SQc * Dc;       // 1,048,576
  const size_t nW     = (size_t)Dc * Dc;             // 262,144
  const size_t nScore = (size_t)Bc * Hc * SQc * SKc; // 8,388,608
  const size_t nP     = (size_t)Bc * SQc * Hc * Dc;  // 8,388,608

  // Workspace bump allocator (256B aligned)
  char* w = (char*)d_ws;
  size_t off = 0;
  auto alloc = [&](size_t bytes) -> void* {
    void* p = w + off;
    off = (off + bytes + 255) & ~(size_t)255;
    return p;
  };
  __bf16* q_bf  = (__bf16*)alloc(nQKV * 2);
  __bf16* k_bf  = (__bf16*)alloc(nQKV * 2);
  __bf16* v_bf  = (__bf16*)alloc(nQKV * 2);
  __bf16* WqT   = (__bf16*)alloc(nW * 2);
  __bf16* WkT   = (__bf16*)alloc(nW * 2);
  __bf16* WvT   = (__bf16*)alloc(nW * 2);
  __bf16* WoT   = (__bf16*)alloc(nW * 2);
  __bf16* Wr_bf = (__bf16*)alloc(nW * 2);
  float*  qh_f  = (float*) alloc(nQKV * 4);
  __bf16* kh_bf = (__bf16*)alloc(nQKV * 2);
  __bf16* vh_bf = (__bf16*)alloc(nQKV * 2);
  __bf16* vT_bf = (__bf16*)alloc(nQKV * 2);
  __bf16* qu_bf = (__bf16*)alloc(nQKV * 2);
  __bf16* qv_bf = (__bf16*)alloc(nQKV * 2);
  __bf16* p_bf  = (__bf16*)alloc(nP * 2);
  float*  cadd  = (float*) alloc((size_t)Bc * SQc * Hc * 4);
  float*  score = (float*) alloc(nScore * 4);
  __bf16* at_bf = (__bf16*)alloc(nScore * 2);
  __bf16* ao_bf = (__bf16*)alloc(nQKV * 2);
  float*  o_f   = (float*) alloc(nQKV * 4);

  float* out_norm = (float*)d_out;              // [B,SQ,D]
  float* out_attn = out_norm + nQKV;            // [B,H,SQ,SK]

  const int CB = 256;
  const int gQKV = (int)(nQKV / CB);            // 4096
  const int gW   = (int)(nW / CB);              // 1024

  // 1) inputs -> bf16; weights -> transposed bf16 (Bt layout); Wr row-major bf16
  cvt_f32_bf16<<<gQKV, CB, 0, stream>>>(q_in, q_bf, (int)nQKV);
  cvt_f32_bf16<<<gQKV, CB, 0, stream>>>(k_in, k_bf, (int)nQKV);
  cvt_f32_bf16<<<gQKV, CB, 0, stream>>>(v_in, v_bf, (int)nQKV);
  transpose_cvt_w<<<dim3(16, 16), CB, 0, stream>>>(Wq, WqT);
  transpose_cvt_w<<<dim3(16, 16), CB, 0, stream>>>(Wk, WkT);
  transpose_cvt_w<<<dim3(16, 16), CB, 0, stream>>>(Wv, WvT);
  transpose_cvt_w<<<dim3(16, 16), CB, 0, stream>>>(Wo, WoT);
  cvt_f32_bf16<<<gW, CB, 0, stream>>>(Wr, Wr_bf, (int)nW);

  const int M2 = Bc * SQc;  // 2048
  // 2) projections: qh (f32), kh (bf16), vh (bf16)
  gemm_bf16_wmma<<<dim3(8, 32, 1), 128, 0, stream>>>(
      q_bf, WqT, bq, nullptr, qh_f, 0, M2, Dc, Dc, Dc, Dc, Dc, 1, 0, 0, 0, 0, 0, 0);
  gemm_bf16_wmma<<<dim3(8, 32, 1), 128, 0, stream>>>(
      k_bf, WkT, bk, nullptr, kh_bf, 1, M2, Dc, Dc, Dc, Dc, Dc, 1, 0, 0, 0, 0, 0, 0);
  gemm_bf16_wmma<<<dim3(8, 32, 1), 128, 0, stream>>>(
      v_bf, WvT, bv, nullptr, vh_bf, 1, M2, Dc, Dc, Dc, Dc, Dc, 1, 0, 0, 0, 0, 0, 0);

  // 3) vh -> vT (Bt layout for attn@V), qu/qv (bias-added bf16)
  transpose_v_kernel<<<gQKV, CB, 0, stream>>>(vh_bf, vT_bf, (int)nQKV);
  addbias_uv<<<gQKV, CB, 0, stream>>>(qh_f, ub, vb, qu_bf, qv_bf, (int)nQKV);

  // 4) p[b,q,h,:] = qv_h @ Wr_h^T  (per-head batched, K=64, bf16 out)
  gemm_bf16_wmma<<<dim3(8, 32, Hc), 128, 0, stream>>>(
      qv_bf, Wr_bf, nullptr, nullptr, p_bf, 1, M2, Dc, DHc, Dc, Dc, Hc * Dc,
      Hc, 0, DHc, 0, DHc, 0, Dc);

  // 5) br correction term
  brdot_kernel<<<(Bc * SQc * Hc + CB - 1) / CB, CB, 0, stream>>>(
      qh_f, vb, br, cadd, Bc * SQc * Hc);

  // 6) ac = qu_h @ kh_h^T  (batched over (b,h), K=64)
  gemm_bf16_wmma<<<dim3(8, 8, Bc * Hc), 128, 0, stream>>>(
      qu_bf, kh_bf, nullptr, nullptr, score, 0, SQc, SKc, DHc, Dc, Dc, SKc,
      Hc, (long)SQc * Dc, DHc, (long)SKc * Dc, DHc,
      (long)Hc * SQc * SKc, (long)SQc * SKc);

  // 7) bd += p . pos_emb  (streams the 512MB pos_emb exactly once)
  bd_kernel<<<dim3(SKc / 64, SQc), 256, 0, stream>>>(p_bf, pos, score);

  // 8) masked softmax -> attn weights (f32 to d_out) + bf16 copy
  softmax_kernel<<<Bc * Hc * SQc, CB, 0, stream>>>(score, cadd, kpm, amask,
                                                   out_attn, at_bf);

  // 9) attn @ V  (batched over (b,h), N=64, bf16 out)
  gemm_bf16_wmma<<<dim3(1, 8, Bc * Hc), 128, 0, stream>>>(
      at_bf, vT_bf, nullptr, nullptr, ao_bf, 1, SQc, DHc, SKc, SKc, SKc, Dc,
      Hc, (long)Hc * SQc * SKc, (long)SQc * SKc,
      (long)Hc * DHc * SKc, (long)DHc * SKc, (long)SQc * Dc, DHc);

  // 10) output projection + bias + residual
  gemm_bf16_wmma<<<dim3(8, 32, 1), 128, 0, stream>>>(
      ao_bf, WoT, bo, q_in, o_f, 0, M2, Dc, Dc, Dc, Dc, Dc, 1, 0, 0, 0, 0, 0, 0);

  // 11) layernorm -> normed output
  layernorm_kernel<<<M2, CB, 0, stream>>>(o_f, lg, lb, out_norm);
}
